// G_GAT_84516366451130
// MI455X (gfx1250) — matrix operand order repaired and checked by reading, not compiled, not run on hardware
//
#include <hip/hip_runtime.h>
#include <hip/hip_bf16.h>

#define NHEADS 33
#define COUT   16
#define FIN    128
#define NCOLS  (NHEADS * COUT)   // 528
#define LDST   136               // bf16 LDS row stride (128 + 8 pad -> conflict-free)
#define NEGSLOPE 0.2f

typedef __attribute__((ext_vector_type(16))) __bf16 v16bf;
typedef __attribute__((ext_vector_type(8)))  float  v8f;

union BF16x16 { v16bf v; unsigned short u[16]; };

__device__ __forceinline__ unsigned short f2bf(float f) {
    unsigned int x = __float_as_uint(f);
    unsigned int r = (x + 0x7FFFu + ((x >> 16) & 1u)) >> 16;   // RNE
    return (unsigned short)r;
}

__device__ __forceinline__ void atomicMaxFloat(float* addr, float val) {
    if (val >= 0.0f) atomicMax((int*)addr, __float_as_int(val));
    else             atomicMin((unsigned int*)addr, __float_as_uint(val));
}

// ---------------------------------------------------------------------------
// Pipelined head loop, instantiated twice (FULL: unguarded stores; tail:
// per-row guards).  Selected once per wave by a scalar branch so the common
// path carries zero guard logic inside the loop.
// ---------------------------------------------------------------------------
template <bool FULL>
__device__ __forceinline__ void gemm_head_loop(
    const BF16x16 (&afrag)[4],
    const unsigned short* __restrict__ bcol0,
    float* __restrict__ rowp,
    int rowTop)
{
    uint4 bufA[8], bufB[8];

    auto loadB = [&](int j, uint4* buf) {
#pragma unroll
        for (int t = 0; t < 4; ++t) {
            const unsigned short* bp = bcol0 + 16 * j * LDST + t * 32;
            buf[2 * t]     = ((const uint4*)bp)[0];   // ds_load_b128
            buf[2 * t + 1] = ((const uint4*)bp)[2];
        }
    };

    auto computeStore = [&](int j, const uint4* buf) {
        v8f acc = {};
#pragma unroll
        for (int t = 0; t < 4; ++t) {
            BF16x16 b;
            ((uint4*)b.u)[0] = buf[2 * t];
            ((uint4*)b.u)[1] = buf[2 * t + 1];
            acc = __builtin_amdgcn_wmma_f32_16x16x32_bf16(
                false, afrag[t].v, false, b.v, (short)0, acc, false, false);
        }
        float* p = rowp + j * COUT;
        if (FULL) {
#pragma unroll
            for (int r = 0; r < 8; ++r) p[(size_t)r * NCOLS] = acc[r];
        } else {
#pragma unroll
            for (int r = 0; r < 8; ++r)
                if (r < rowTop) p[(size_t)r * NCOLS] = acc[r];
        }
    };

    loadB(0, bufA);
    for (int j = 0; j < NHEADS - 1; j += 2) {
        loadB(j + 1, bufB);
        computeStore(j, bufA);
        loadB(j + 2, bufA);          // j+2 <= 32 always valid here
        computeStore(j + 1, bufB);
    }
    computeStore(NHEADS - 1, bufA);  // j = 32
}

// ---------------------------------------------------------------------------
// 1) h = x @ W  via bf16 WMMA.  One block = 128 rows, 8 waves, wave = 16-row
//    strip across all 33 heads.  W staged in LDS as bf16, transposed+padded.
// ---------------------------------------------------------------------------
__global__ __launch_bounds__(256) void gat_gemm_kernel(
    const float* __restrict__ x,     // [N,128]
    const float* __restrict__ W,     // [128,528] row-major
    float* __restrict__ hbuf,        // [N,528]
    int N)
{
    extern __shared__ unsigned short ldsW[];   // [528][LDST] bf16: ldsW[col*LDST + k]
    const int tid = threadIdx.x;

    // stage W -> LDS (coalesced global read, transposed bf16 store)
    for (int idx = tid; idx < FIN * NCOLS; idx += 256) {
        int k   = idx / NCOLS;
        int col = idx - k * NCOLS;
        ldsW[col * LDST + k] = f2bf(W[idx]);
    }
    __syncthreads();

    const int wave = tid >> 5;
    const int lane = tid & 31;
    const int m16  = lane & 15;
    const int half = lane >> 4;
    const int row0 = blockIdx.x * 128 + wave * 16;

    // wave-uniform tail predicate
    const bool fullTile =
        (bool)__builtin_amdgcn_readfirstlane((int)(row0 + 16 <= N));

    int rowc = row0 + m16;
    if (rowc > N - 1) rowc = N - 1;               // clamp loads; stores guarded
    const float* xr = x + (size_t)rowc * FIN;

    // A fragments: 16x32 bf16 per k-step.  ISA layout:
    //   elems 0..7  -> K = 32t + half*8 + i
    //   elems 8..15 -> K = 32t + 16 + half*8 + i
    BF16x16 afrag[4];
#pragma unroll
    for (int t = 0; t < 4; ++t) {
        const float* lo = xr + t * 32 + half * 8;
        float4 l0 = *(const float4*)(lo);
        float4 l1 = *(const float4*)(lo + 4);
        float4 h0 = *(const float4*)(lo + 16);
        float4 h1 = *(const float4*)(lo + 20);
        afrag[t].u[0]  = f2bf(l0.x); afrag[t].u[1]  = f2bf(l0.y);
        afrag[t].u[2]  = f2bf(l0.z); afrag[t].u[3]  = f2bf(l0.w);
        afrag[t].u[4]  = f2bf(l1.x); afrag[t].u[5]  = f2bf(l1.y);
        afrag[t].u[6]  = f2bf(l1.z); afrag[t].u[7]  = f2bf(l1.w);
        afrag[t].u[8]  = f2bf(h0.x); afrag[t].u[9]  = f2bf(h0.y);
        afrag[t].u[10] = f2bf(h0.z); afrag[t].u[11] = f2bf(h0.w);
        afrag[t].u[12] = f2bf(h1.x); afrag[t].u[13] = f2bf(h1.y);
        afrag[t].u[14] = f2bf(h1.z); afrag[t].u[15] = f2bf(h1.w);
    }

    // per-lane store base: row (row0 + 8*half + r), col (j*16 + m16)
    float* rowp = hbuf + (size_t)(row0 + 8 * half) * NCOLS + m16;
    const int rowTop = N - (row0 + 8 * half);     // #valid r in this half

    // B fragment 32x16: lane holds col n=m16, elems 0..7 -> K = 32t + half*8 + i, elems 8..15 -> K = 32t + 16 + half*8 + i
    const unsigned short* bcol0 = ldsW + m16 * LDST + half * 8;

    if (fullTile) gemm_head_loop<true >(afrag, bcol0, rowp, rowTop);
    else          gemm_head_loop<false>(afrag, bcol0, rowp, rowTop);
}

// ---------------------------------------------------------------------------
// 2) per-(node,head) attention logits (hbuf is L2-resident -> cheap pass)
// ---------------------------------------------------------------------------
__global__ void gat_att_kernel(const float* __restrict__ hbuf,
                               const float* __restrict__ att_src,
                               const float* __restrict__ att_dst,
                               float* __restrict__ a_src,
                               float* __restrict__ a_dst, int N)
{
    int idx = blockIdx.x * 256 + threadIdx.x;     // n*H + h
    if (idx >= N * NHEADS) return;
    int hh = idx % NHEADS;
    const float* hp = hbuf + (size_t)idx * COUT;
    float s = 0.f, d = 0.f;
#pragma unroll
    for (int c = 0; c < COUT; ++c) {
        float v = hp[c];
        s += v * att_src[hh * COUT + c];
        d += v * att_dst[hh * COUT + c];
    }
    a_src[idx] = s;
    a_dst[idx] = d;
}

// ---------------------------------------------------------------------------
// 3) init workspace (deterministic: runs every call)
// ---------------------------------------------------------------------------
__global__ void gat_init_kernel(float* __restrict__ m, float* __restrict__ denom,
                                float* __restrict__ accum, int NH, int NHC)
{
    int idx = blockIdx.x * 256 + threadIdx.x;
    if (idx < NH) {
        m[idx]     = __uint_as_float(0xFF800000u);  // -inf
        denom[idx] = 0.f;
    }
    if (idx < NHC) accum[idx] = 0.f;
}

// ---------------------------------------------------------------------------
// 4) edge pass 1: segment max of leaky-relu logits
// ---------------------------------------------------------------------------
__global__ void gat_edge_max_kernel(const int* __restrict__ src,
                                    const int* __restrict__ dst,
                                    const float* __restrict__ a_src,
                                    const float* __restrict__ a_dst,
                                    float* __restrict__ m, int E, int N)
{
    int idx = blockIdx.x * 256 + threadIdx.x;     // e*H + h
    int total = (E + N) * NHEADS;
    if (idx >= total) return;
    int e  = idx / NHEADS;
    int hh = idx - e * NHEADS;
    int s = (e < E) ? src[e] : (e - E);           // self loop
    int d = (e < E) ? dst[e] : (e - E);
    float ev = a_src[s * NHEADS + hh] + a_dst[d * NHEADS + hh];
    ev = (ev > 0.f) ? ev : NEGSLOPE * ev;
    atomicMaxFloat(&m[d * NHEADS + hh], ev);
}

// ---------------------------------------------------------------------------
// 5) edge pass 2: fused softmax numerator + denominator accumulation
// ---------------------------------------------------------------------------
__global__ void gat_edge_acc_kernel(const int* __restrict__ src,
                                    const int* __restrict__ dst,
                                    const float* __restrict__ a_src,
                                    const float* __restrict__ a_dst,
                                    const float* __restrict__ m,
                                    const float* __restrict__ hbuf,
                                    float* __restrict__ denom,
                                    float* __restrict__ accum, int E, int N)
{
    int idx = blockIdx.x * 256 + threadIdx.x;     // e*H + h
    int total = (E + N) * NHEADS;
    if (idx >= total) return;
    int e  = idx / NHEADS;
    int hh = idx - e * NHEADS;
    int s = (e < E) ? src[e] : (e - E);
    int d = (e < E) ? dst[e] : (e - E);
    float ev = a_src[s * NHEADS + hh] + a_dst[d * NHEADS + hh];
    ev = (ev > 0.f) ? ev : NEGSLOPE * ev;
    float ex = __expf(ev - m[d * NHEADS + hh]);
    atomicAdd(&denom[d * NHEADS + hh], ex);
    const float* hp = hbuf + ((size_t)s * NHEADS + hh) * COUT;
    float*       ap = accum + ((size_t)d * NHEADS + hh) * COUT;
#pragma unroll
    for (int c = 0; c < COUT; ++c)
        atomicAdd(&ap[c], ex * hp[c]);
}

// ---------------------------------------------------------------------------
// 6) mean over heads + bias + tanh
// ---------------------------------------------------------------------------
__global__ void gat_out_kernel(const float* __restrict__ accum,
                               const float* __restrict__ denom,
                               const float* __restrict__ bias,
                               float* __restrict__ out, int N)
{
    int idx = blockIdx.x * 256 + threadIdx.x;     // n*C + c
    if (idx >= N * COUT) return;
    int n = idx / COUT;
    int c = idx - n * COUT;
    float s = 0.f;
    for (int hh = 0; hh < NHEADS; ++hh)
        s += accum[((size_t)n * NHEADS + hh) * COUT + c] /
             (denom[n * NHEADS + hh] + 1e-16f);
    s = s * (1.0f / NHEADS) + bias[c];
    out[idx] = tanhf(s);
}

// ---------------------------------------------------------------------------
extern "C" void kernel_launch(void* const* d_in, const int* in_sizes, int n_in,
                              void* d_out, int out_size, void* d_ws, size_t ws_size,
                              hipStream_t stream)
{
    const float* x       = (const float*)d_in[0];
    const int*   ei      = (const int*)  d_in[1];
    const float* W       = (const float*)d_in[2];
    const float* att_src = (const float*)d_in[3];
    const float* att_dst = (const float*)d_in[4];
    const float* bias    = (const float*)d_in[5];

    const int N  = in_sizes[0] / FIN;     // 50000
    const int E  = in_sizes[1] / 2;       // 320000
    const int Et = E + N;

    // workspace partition (256B aligned)
    char* p = (char*)d_ws;
    auto alloc = [&](size_t bytes) {
        char* r = p;
        p += (bytes + 255) & ~(size_t)255;
        return (void*)r;
    };
    float* hbuf  = (float*)alloc((size_t)N * NCOLS  * 4);
    float* a_src = (float*)alloc((size_t)N * NHEADS * 4);
    float* a_dst = (float*)alloc((size_t)N * NHEADS * 4);
    float* m     = (float*)alloc((size_t)N * NHEADS * 4);
    float* denom = (float*)alloc((size_t)N * NHEADS * 4);
    float* accum = (float*)alloc((size_t)N * NCOLS  * 4);

    dim3 blk(256);
    const int    gemmBlocks = (N + 127) / 128;
    const size_t shmem      = (size_t)NCOLS * LDST * 2;   // 143,616 B LDS (2 blocks/WGP)

    gat_gemm_kernel<<<gemmBlocks, blk, shmem, stream>>>(x, W, hbuf, N);

    const int nh = N * NHEADS;
    gat_att_kernel<<<(nh + 255) / 256, blk, 0, stream>>>(hbuf, att_src, att_dst,
                                                         a_src, a_dst, N);

    const int nhc = N * NCOLS;
    gat_init_kernel<<<(nhc + 255) / 256, blk, 0, stream>>>(m, denom, accum, nh, nhc);

    const long long ew = (long long)Et * NHEADS;
    const int eblocks  = (int)((ew + 255) / 256);
    gat_edge_max_kernel<<<eblocks, blk, 0, stream>>>(ei, ei + E, a_src, a_dst,
                                                     m, E, N);
    gat_edge_acc_kernel<<<eblocks, blk, 0, stream>>>(ei, ei + E, a_src, a_dst,
                                                     m, hbuf, denom, accum, E, N);

    const int nc = N * COUT;
    gat_out_kernel<<<(nc + 255) / 256, blk, 0, stream>>>(accum, denom, bias,
                                                         (float*)d_out, N);
}
